// HPUUnifiedMLAImpl_71141838291325
// MI455X (gfx1250) — matrix operand order repaired and checked by reading, not compiled, not run on hardware
//
#include <hip/hip_runtime.h>

#define T_SEQ 2048
#define H_HEADS 16
#define NOPE 128
#define ROPE 64
#define DHEAD 192
#define LORA 512
#define VDIM 128
#define ATTN_SCALE 0.07216878364870323f  // 1/sqrt(192)

typedef __bf16 bf16_t;
typedef __attribute__((ext_vector_type(16))) __bf16 v16bf;
typedef __attribute__((ext_vector_type(8)))  float  v8f;

#ifndef __has_builtin
#define __has_builtin(x) 0
#endif
#if __has_builtin(__builtin_amdgcn_global_load_async_to_lds_b128)
#define HAVE_ASYNC 1
#else
#define HAVE_ASYNC 0
#endif

#if HAVE_ASYNC
typedef int v4i __attribute__((vector_size(16)));
typedef __attribute__((address_space(1))) v4i* gv4i_p;
typedef __attribute__((address_space(3))) v4i* lv4i_p;
#endif

// 16-byte global -> LDS stage. Async (ASYNCcnt-tracked) when available.
__device__ __forceinline__ void cp16(void* lds, const void* g) {
#if HAVE_ASYNC
  __builtin_amdgcn_global_load_async_to_lds_b128((gv4i_p)(void*)g, (lv4i_p)lds,
                                                 0, 0);
#else
  *(uint4*)lds = *(const uint4*)g;
#endif
}

__device__ __forceinline__ void wait_async_all() {
#if HAVE_ASYNC
#if __has_builtin(__builtin_amdgcn_s_wait_asynccnt)
  __builtin_amdgcn_s_wait_asynccnt(0);
#else
  asm volatile("s_wait_asynccnt 0x0" ::: "memory");
#endif
#endif
}

// ---------------- WMMA fragment helpers (CDNA5 wave32 layouts, ISA 7.12.2) ---

// A matrix 16x32 bf16, row-major with row stride ld. Two contiguous 16B runs.
__device__ __forceinline__ v16bf load_A16x32(const bf16_t* base, int ld) {
  const int lane = threadIdx.x & 31;
  const int m = lane & 15, hi = lane >> 4;
  const bf16_t* row = base + m * ld;
  v16bf a;
#pragma unroll
  for (int i = 0; i < 8; ++i) a[i] = row[hi * 8 + i];
#pragma unroll
  for (int i = 0; i < 8; ++i) a[8 + i] = row[16 + hi * 8 + i];
  return a;
}

// B matrix 32x16 bf16 from an N-major (transposed) tile: B[k][n] = base[n*ld+k].
// lanes 0-15: col n, K=0..7,16..23; lanes 16-31: col n, K=8..15,24..31. Two 16B runs.
__device__ __forceinline__ v16bf load_B_nxk(const bf16_t* base, int ld) {
  const int lane = threadIdx.x & 31;
  const int n = lane & 15, hi = lane >> 4;
  const bf16_t* p = base + n * ld + hi * 8;
  v16bf b;
#pragma unroll
  for (int e = 0; e < 8; ++e) b[e] = p[e];
#pragma unroll
  for (int e = 0; e < 8; ++e) b[8 + e] = p[16 + e];
  return b;
}

__device__ __forceinline__ v8f wmma_bf16(v16bf a, v16bf b, v8f c) {
  return __builtin_amdgcn_wmma_f32_16x16x32_bf16(false, a, false, b, (short)0, c,
                                                 false, false);
}

// ---------------- Kernel 1: convert / reshape / transpose to bf16 ------------

__global__ void __launch_bounds__(256)
prep_kernel(const float* __restrict__ query, const float* __restrict__ kc,
            const float* __restrict__ kpe, const float* __restrict__ wkvb,
            const float* __restrict__ wuv, bf16_t* __restrict__ q_bf,
            bf16_t* __restrict__ kc_bf, bf16_t* __restrict__ kcT,
            bf16_t* __restrict__ WkT, bf16_t* __restrict__ wuvT,
            bf16_t* __restrict__ Khead) {
  const int task = blockIdx.y;
  const long stride = (long)gridDim.x * blockDim.x;
  const long i0 = (long)blockIdx.x * blockDim.x + threadIdx.x;
  if (task == 0) {
    const long N = (long)T_SEQ * H_HEADS * DHEAD;
    for (long i = i0; i < N; i += stride) q_bf[i] = (bf16_t)query[i];
  } else if (task == 1) {
    const long N = (long)T_SEQ * LORA;
    for (long i = i0; i < N; i += stride) kc_bf[i] = (bf16_t)kc[i];
  } else if (task == 2) {
    // WkT[col][l] = wkvb[l][(col>>7)*256 + (col&127)]  (N-major W_k, nope cols)
    const long N = (long)(H_HEADS * NOPE) * LORA;
    for (long i = i0; i < N; i += stride) {
      const int l = (int)(i & (LORA - 1));
      const int col = (int)(i >> 9);
      WkT[i] = (bf16_t)wkvb[(long)l * (H_HEADS * (NOPE + VDIM)) +
                            (col >> 7) * (NOPE + VDIM) + (col & (NOPE - 1))];
    }
  } else if (task == 3) {
    // wuvT[h][v][l] = wuv[h][l][v]
    const long N = (long)H_HEADS * VDIM * LORA;
    for (long i = i0; i < N; i += stride) {
      const int l = (int)(i & (LORA - 1));
      const int v = (int)((i >> 9) & (VDIM - 1));
      const int h = (int)(i >> 16);
      wuvT[i] = (bf16_t)wuv[((long)h * LORA + l) * VDIM + v];
    }
  } else if (task == 4) {
    // rope: Khead[h][t][128+r] = kpe[t][r]
    const long N = (long)T_SEQ * H_HEADS * ROPE;
    for (long i = i0; i < N; i += stride) {
      const int r = (int)(i & (ROPE - 1));
      const int h = (int)((i >> 6) & (H_HEADS - 1));
      const int t = (int)(i >> 10);
      Khead[(long)h * T_SEQ * DHEAD + (long)t * DHEAD + NOPE + r] =
          (bf16_t)kpe[(long)t * ROPE + r];
    }
  } else {
    // kcT[l][t] = kc[t][l]  (V operand, N-major)
    const long N = (long)LORA * T_SEQ;
    for (long i = i0; i < N; i += stride) {
      const int t = (int)(i & (T_SEQ - 1));
      const int l = (int)(i >> 11);
      kcT[i] = (bf16_t)kc[(long)t * LORA + l];
    }
  }
}

// ---------------- Kernel 2: K_nope = k_c @ W_k (bf16 WMMA, async staging) ----
// M=2048, N=2048, K=512. Block = 4 waves, 64x64 tile.

__global__ void __launch_bounds__(128)
gemm_knope(const bf16_t* __restrict__ A,   // kc_bf (T x 512) row-major
           const bf16_t* __restrict__ BT,  // WkT (2048 x 512) N-major
           bf16_t* __restrict__ Khead) {   // (H,T,192), d in [0,128)
  __shared__ bf16_t Asl[64 * 32];
  __shared__ bf16_t Bsl[64 * 32];
  __shared__ bf16_t Csl[64 * 64];
  const int tid = threadIdx.x;
  const int wave = tid >> 5;
  const int m0 = blockIdx.x * 64;
  const int n0 = blockIdx.y * 64;
  const v8f vzero = {0.f, 0.f, 0.f, 0.f, 0.f, 0.f, 0.f, 0.f};
  v8f acc[4] = {vzero, vzero, vzero, vzero};

  for (int k0 = 0; k0 < LORA; k0 += 32) {
    __syncthreads();
#pragma unroll
    for (int it = 0; it < 2; ++it) {  // 64 rows x 4 x 16B per tile
      const int seg = tid + it * 128;
      const int r = seg >> 2, c = seg & 3;
      cp16(Asl + r * 32 + c * 8, A + (long)(m0 + r) * LORA + k0 + c * 8);
      cp16(Bsl + r * 32 + c * 8, BT + (long)(n0 + r) * LORA + k0 + c * 8);
    }
    wait_async_all();
    __syncthreads();
    const v16bf a = load_A16x32(Asl + wave * 16 * 32, 32);
#pragma unroll
    for (int g = 0; g < 4; ++g) {
      const v16bf b = load_B_nxk(Bsl + g * 16 * 32, 32);
      acc[g] = wmma_bf16(a, b, acc[g]);
    }
  }

  // stage C through LDS, then coalesced 16B stores into Khead
  const int lane = tid & 31;
  const int n = lane & 15, hi = lane >> 4;
#pragma unroll
  for (int g = 0; g < 4; ++g)
#pragma unroll
    for (int r = 0; r < 8; ++r)
      Csl[(wave * 16 + r + 8 * hi) * 64 + g * 16 + n] = (bf16_t)acc[g][r];
  __syncthreads();
  const int h = n0 >> 7, d0 = n0 & (NOPE - 1);
  bf16_t* outp = Khead + (long)h * T_SEQ * DHEAD + d0;
#pragma unroll
  for (int it = 0; it < 2; ++it) {  // 64 rows x 4 x 16B
    const int seg = tid + it * 128;
    const int r = seg >> 2, c = seg & 3;
    *(uint4*)(outp + (long)(m0 + r) * DHEAD + c * 8) =
        *(const uint4*)(Csl + r * 64 + c * 8);
  }
}

// ---------------- Kernel 3: causal flash attention, V = k_c (dv=512) ---------
// Block = 4 waves / head / 16-row q tile. Double-buffered async K/V staging.

__device__ __forceinline__ void stage_kv(const bf16_t* Kg, const bf16_t* kcT,
                                         int s0, bf16_t* Ksb, bf16_t* Vsb,
                                         int tid) {
#pragma unroll
  for (int it = 0; it < 6; ++it) {  // K: 32 rows x 24 x 16B
    const int seg = tid + it * 128;
    const int r = seg / 24, c = seg % 24;
    cp16(Ksb + r * DHEAD + c * 8, Kg + (long)(s0 + r) * DHEAD + c * 8);
  }
#pragma unroll
  for (int it = 0; it < 16; ++it) {  // V^T: 512 rows x 4 x 16B (32 s-cols)
    const int seg = tid + it * 128;
    const int r = seg >> 2, c = seg & 3;
    cp16(Vsb + r * 32 + c * 8, kcT + (long)r * T_SEQ + s0 + c * 8);
  }
}

__global__ void __launch_bounds__(128)
flash_kernel(const bf16_t* __restrict__ q_bf,   // (T,H,192)
             const bf16_t* __restrict__ Khead,  // (H,T,192)
             const bf16_t* __restrict__ kcT,    // (512,T)  N-major V
             bf16_t* __restrict__ ol) {         // (H,T,512)
  __shared__ bf16_t Qs[16 * DHEAD];       // 6 KB
  __shared__ bf16_t Ks[2][32 * DHEAD];    // 2 x 12 KB
  __shared__ bf16_t Vs[2][LORA * 32];     // 2 x 32 KB (N-major: [l][s])
  __shared__ bf16_t Ps[4][16 * 32];       // 4 KB

  const int tid = threadIdx.x;
  const int wave = tid >> 5;
  const int lane = tid & 31;
  const int n = lane & 15, hi = lane >> 4;
  const int t0 = blockIdx.x * 16;
  const int h = blockIdx.y;
  const bf16_t* Kg = Khead + (long)h * T_SEQ * DHEAD;

  // prologue: stage Q tile + chunk 0
#pragma unroll
  for (int it = 0; it < 3; ++it) {  // Q: 16 rows x 24 x 16B
    const int seg = tid + it * 128;
    const int r = seg / 24, c = seg % 24;
    cp16(Qs + r * DHEAD + c * 8,
         q_bf + ((long)(t0 + r) * H_HEADS + h) * DHEAD + c * 8);
  }
  stage_kv(Kg, kcT, 0, Ks[0], Vs[0], tid);

  const v8f vzero = {0.f, 0.f, 0.f, 0.f, 0.f, 0.f, 0.f, 0.f};
  v8f o[8] = {vzero, vzero, vzero, vzero, vzero, vzero, vzero, vzero};
  float m_i[8], l_i[8];
#pragma unroll
  for (int r = 0; r < 8; ++r) { m_i[r] = -1e30f; l_i[r] = 0.f; }

  const int nchunks = (t0 + 15) / 32 + 1;
  for (int ch = 0; ch < nchunks; ++ch) {
    const int s0 = ch * 32;
    wait_async_all();   // own async loads for chunk ch have landed
    __syncthreads();    // everyone's landed; everyone done with other buffer
    if (ch + 1 < nchunks)
      stage_kv(Kg, kcT, s0 + 32, Ks[(ch + 1) & 1], Vs[(ch + 1) & 1], tid);
    const bf16_t* Ksb = Ks[ch & 1];
    const bf16_t* Vsb = Vs[ch & 1];

    // S(16x32) = Q x K^T, two 16x16 D-frags
    v8f sf0 = vzero, sf1 = vzero;
#pragma unroll
    for (int kk = 0; kk < 6; ++kk) {
      const v16bf a = load_A16x32(Qs + kk * 32, DHEAD);
      const v16bf b0 = load_B_nxk(Ksb + kk * 32, DHEAD);
      const v16bf b1 = load_B_nxk(Ksb + 16 * DHEAD + kk * 32, DHEAD);
      sf0 = wmma_bf16(a, b0, sf0);
      sf1 = wmma_bf16(a, b1, sf1);
    }

    // scale + causal mask + online softmax (rows r+8*hi, cols n / 16+n)
    float scale_r[8];
#pragma unroll
    for (int r = 0; r < 8; ++r) {
      const int trow = t0 + r + 8 * hi;
      float v0 = sf0[r] * ATTN_SCALE;
      float v1 = sf1[r] * ATTN_SCALE;
      if (s0 + n > trow) v0 = -1e30f;
      if (s0 + 16 + n > trow) v1 = -1e30f;
      float mx = fmaxf(v0, v1);
#pragma unroll
      for (int msk = 8; msk >= 1; msk >>= 1)
        mx = fmaxf(mx, __shfl_xor(mx, msk, 32));
      const float m_new = fmaxf(m_i[r], mx);
      const float p0 = __expf(v0 - m_new);
      const float p1 = __expf(v1 - m_new);
      sf0[r] = p0;
      sf1[r] = p1;
      float sum = p0 + p1;
#pragma unroll
      for (int msk = 8; msk >= 1; msk >>= 1) sum += __shfl_xor(sum, msk, 32);
      const float sc = __expf(m_i[r] - m_new);
      l_i[r] = l_i[r] * sc + sum;
      m_i[r] = m_new;
      scale_r[r] = sc;
    }

    // stage P (16x32) through LDS to A-fragment layout
    bf16_t* pb = Ps[wave];
#pragma unroll
    for (int r = 0; r < 8; ++r) {
      pb[(r + 8 * hi) * 32 + n] = (bf16_t)sf0[r];
      pb[(r + 8 * hi) * 32 + 16 + n] = (bf16_t)sf1[r];
    }
    __syncthreads();

    // O(16x128 slice) = O*diag(scale) + P(16x32) @ V(32x128 slice)
    const v16bf pa = load_A16x32(pb, 32);
#pragma unroll
    for (int g = 0; g < 8; ++g) {
#pragma unroll
      for (int r = 0; r < 8; ++r) o[g][r] *= scale_r[r];
      const v16bf b = load_B_nxk(Vsb + (wave * 128 + g * 16) * 32, 32);
      o[g] = wmma_bf16(pa, b, o[g]);
    }
  }

  // epilogue: normalize, stage through (now-dead) Vs[0], coalesced stores
  __syncthreads();
  bf16_t* Cs = &Vs[0][0];  // 16 x 512 bf16 staging
#pragma unroll
  for (int r = 0; r < 8; ++r) {
    const float inv = 1.0f / l_i[r];
#pragma unroll
    for (int g = 0; g < 8; ++g)
      Cs[(r + 8 * hi) * LORA + wave * 128 + g * 16 + n] =
          (bf16_t)(o[g][r] * inv);
  }
  __syncthreads();
  bf16_t* olh = ol + (long)h * T_SEQ * LORA;
#pragma unroll
  for (int it = 0; it < 8; ++it) {  // 16 rows x 64 x 16B
    const int seg = tid + it * 128;
    const int r = seg >> 6, c = seg & 63;
    *(uint4*)(olh + (long)(t0 + r) * LORA + c * 8) =
        *(const uint4*)(Cs + r * LORA + c * 8);
  }
}

// ---------------- Kernel 4: out = out_latent @ w_uv (per head, f32 out) ------

__global__ void __launch_bounds__(128)
gemm_out(const bf16_t* __restrict__ ol,    // (H,T,512) row-major
         const bf16_t* __restrict__ wuvT,  // (H,128,512) N-major
         float* __restrict__ out) {        // (T, H*128)
  __shared__ bf16_t Asl[64 * 32];
  __shared__ bf16_t Bsl[64 * 32];
  const int tid = threadIdx.x;
  const int wave = tid >> 5;
  const int m0 = blockIdx.x * 64;
  const int n0 = blockIdx.y * 64;
  const int h = blockIdx.z;
  const bf16_t* A = ol + (long)h * T_SEQ * LORA;
  const bf16_t* BT = wuvT + (long)h * VDIM * LORA;
  const v8f vzero = {0.f, 0.f, 0.f, 0.f, 0.f, 0.f, 0.f, 0.f};
  v8f acc[4] = {vzero, vzero, vzero, vzero};

  for (int k0 = 0; k0 < LORA; k0 += 32) {
    __syncthreads();
#pragma unroll
    for (int it = 0; it < 2; ++it) {
      const int seg = tid + it * 128;
      const int r = seg >> 2, c = seg & 3;
      cp16(Asl + r * 32 + c * 8, A + (long)(m0 + r) * LORA + k0 + c * 8);
      cp16(Bsl + r * 32 + c * 8, BT + (long)(n0 + r) * LORA + k0 + c * 8);
    }
    wait_async_all();
    __syncthreads();
    const v16bf a = load_A16x32(Asl + wave * 16 * 32, 32);
#pragma unroll
    for (int g = 0; g < 4; ++g) {
      const v16bf b = load_B_nxk(Bsl + g * 16 * 32, 32);
      acc[g] = wmma_bf16(a, b, acc[g]);
    }
  }

  const int lane = tid & 31;
  const int n = lane & 15, hi = lane >> 4;
#pragma unroll
  for (int g = 0; g < 4; ++g)
#pragma unroll
    for (int r = 0; r < 8; ++r) {
      const int t = m0 + wave * 16 + r + 8 * hi;
      out[(long)t * (H_HEADS * VDIM) + h * VDIM + n0 + g * 16 + n] = acc[g][r];
    }
}

// ---------------- host launch ------------------------------------------------

extern "C" void kernel_launch(void* const* d_in, const int* in_sizes, int n_in,
                              void* d_out, int out_size, void* d_ws,
                              size_t ws_size, hipStream_t stream) {
  (void)in_sizes; (void)n_in; (void)out_size; (void)ws_size;
  const float* query = (const float*)d_in[0];
  const float* kc = (const float*)d_in[1];
  const float* kpe = (const float*)d_in[2];
  const float* wkvb = (const float*)d_in[3];
  const float* wuv = (const float*)d_in[4];
  float* out = (float*)d_out;

  char* p = (char*)d_ws;
  auto alloc = [&](size_t bytes) {
    char* r = p;
    p += (bytes + 255) & ~(size_t)255;
    return r;
  };
  bf16_t* q_bf = (bf16_t*)alloc((size_t)T_SEQ * H_HEADS * DHEAD * 2);
  bf16_t* kc_bf = (bf16_t*)alloc((size_t)T_SEQ * LORA * 2);
  bf16_t* kcT = (bf16_t*)alloc((size_t)LORA * T_SEQ * 2);
  bf16_t* WkT = (bf16_t*)alloc((size_t)(H_HEADS * NOPE) * LORA * 2);
  bf16_t* wuvT = (bf16_t*)alloc((size_t)H_HEADS * VDIM * LORA * 2);
  bf16_t* Khead = (bf16_t*)alloc((size_t)H_HEADS * T_SEQ * DHEAD * 2);
  bf16_t* ol = (bf16_t*)alloc((size_t)H_HEADS * T_SEQ * LORA * 2);

  prep_kernel<<<dim3(256, 6), 256, 0, stream>>>(query, kc, kpe, wkvb, wuv, q_bf,
                                                kc_bf, kcT, WkT, wuvT, Khead);
  gemm_knope<<<dim3(32, 32), 128, 0, stream>>>(kc_bf, WkT, Khead);
  flash_kernel<<<dim3(T_SEQ / 16, H_HEADS), 128, 0, stream>>>(q_bf, Khead, kcT,
                                                              ol);
  gemm_out<<<dim3(32, 2, H_HEADS), 128, 0, stream>>>(ol, wuvT, out);
}
